// ExaoneMoEDecoderLayer_26620207301234
// MI455X (gfx1250) — matrix operand order repaired and checked by reading, not compiled, not run on hardware
//
#include <hip/hip_runtime.h>

// ---------------------------------------------------------------------------
// Types for CDNA5 WMMA (wave32)
// ---------------------------------------------------------------------------
typedef __bf16 bf16;
typedef __attribute__((ext_vector_type(16))) __bf16 v16bf;
typedef __attribute__((ext_vector_type(8)))  __bf16 v8bf;
typedef __attribute__((ext_vector_type(8)))  float   v8f;

// ---------------------------------------------------------------------------
// Problem constants
// ---------------------------------------------------------------------------
constexpr int Bc   = 2;
constexpr int Sc   = 1024;
constexpr int Hc   = 2048;
constexpr int NHc  = 16;
constexpr int NKVc = 8;
constexpr int DHc  = 128;
constexpr int Ec   = 16;
constexpr int Kc   = 4;          // top-k
constexpr int Ic   = 768;
constexpr int I2c  = 1536;
constexpr int ISHc = 768;
constexpr int ISH2c= 1536;
constexpr int Tc   = Bc * Sc;                 // 2048 tokens
constexpr int QNc  = (NHc + 2 * NKVc) * DHc;  // 4096
constexpr int NHDc = NHc * DHc;               // 2048
constexpr int TKc  = Tc * Kc;                 // 8192 expert-token slots

__device__ __forceinline__ bf16 f2bf(float f) {
  unsigned u = __float_as_uint(f);
  u += 0x7fffu + ((u >> 16) & 1u);            // round-to-nearest-even
  unsigned short h = (unsigned short)(u >> 16);
  return __builtin_bit_cast(bf16, h);
}
__device__ __forceinline__ float bf2f(bf16 b) {
  unsigned short h = __builtin_bit_cast(unsigned short, b);
  return __uint_as_float(((unsigned)h) << 16);
}
__device__ __forceinline__ v16bf concat8(v8bf lo, v8bf hi) {
  return __builtin_shufflevector(lo, hi, 0,1,2,3,4,5,6,7,8,9,10,11,12,13,14,15);
}
__device__ __forceinline__ v16bf load_afrag(const bf16* rowk0, int half) {
  // A(16x32) fragment: per lane one contiguous run of 16 at +16*half
  const bf16* p = rowk0 + 16 * half;
  v8bf lo = *(const v8bf*)p;
  v8bf hi = *(const v8bf*)(p + 8);
  return concat8(lo, hi);
}
__device__ __forceinline__ v8f wmma_bf16(v16bf a, v16bf b, v8f c) {
  return __builtin_amdgcn_wmma_f32_16x16x32_bf16(false, a, false, b, (short)0, c, false, false);
}

// ---------------------------------------------------------------------------
// Weight convert + transpose:  W[e][k][n] (f32) -> Wt[e][n][k] (bf16)
// ---------------------------------------------------------------------------
__global__ void conv_t_kernel(const float* __restrict__ W, bf16* __restrict__ Wt,
                              int Kd, int Nd, int Ed) {
  size_t i   = (size_t)blockIdx.x * 256 + threadIdx.x;
  size_t per = (size_t)Kd * Nd;
  size_t tot = per * Ed;
  if (i >= tot) return;
  size_t e   = i / per;
  size_t rem = i - e * per;
  int n = (int)(rem % Nd);
  int k = (int)(rem / Nd);
  Wt[e * per + (size_t)n * Kd + k] = f2bf(W[i]);
}

// ---------------------------------------------------------------------------
// RMSNorm (fp32 in -> bf16 out), one block per token
// ---------------------------------------------------------------------------
__global__ __launch_bounds__(256) void rmsnorm_kernel(const float* __restrict__ X,
                                                      const float* __restrict__ W,
                                                      bf16* __restrict__ O, int Hd) {
  int t = blockIdx.x;
  const float* x = X + (size_t)t * Hd;
  float ss = 0.f;
  for (int j = threadIdx.x; j < Hd; j += 256) { float v = x[j]; ss += v * v; }
  __shared__ float red[256];
  red[threadIdx.x] = ss; __syncthreads();
  for (int s = 128; s > 0; s >>= 1) {
    if (threadIdx.x < s) red[threadIdx.x] += red[threadIdx.x + s];
    __syncthreads();
  }
  float rs = rsqrtf(red[0] / (float)Hd + 1e-6f);
  for (int j = threadIdx.x; j < Hd; j += 256)
    O[(size_t)t * Hd + j] = f2bf(x[j] * rs * W[j]);
}

// ---------------------------------------------------------------------------
// Register-blocked dense GEMM:
//   C[M][N] (f32, += Res) = A[M][K] (bf16) * Bt[N][K] (bf16)
// Block = 128 threads (4 waves as 2x2); block tile 64(M) x 128(N);
// each wave computes a 32x64 tile = 2x4 WMMA subtiles (8 wmma / k-step,
// 192B of fragments / k-step -> ~2.7x less fragment traffic than 1 tile/wave).
// grid = (N/128, M/64)
// ---------------------------------------------------------------------------
__global__ __launch_bounds__(128) void gemm_bf16_kernel(
    const bf16* __restrict__ A, const bf16* __restrict__ Bt,
    float* __restrict__ C, const float* __restrict__ Res,
    int M, int N, int K) {
  int lane = threadIdx.x & 31;
  int wave = threadIdx.x >> 5;
  int half = lane >> 4;
  int l16  = lane & 15;
  int mw = wave & 1, nw = wave >> 1;
  int m0 = blockIdx.y * 64 + mw * 32;
  int n0 = blockIdx.x * 128 + nw * 64;
  if (m0 >= M || n0 >= N) return;
  const bf16* arow0 = A + (size_t)(m0 + l16) * K;
  const bf16* arow1 = A + (size_t)(m0 + 16 + l16) * K;
  const bf16* brow  = Bt + (size_t)(n0 + l16) * K;
  const size_t bstep = (size_t)16 * K;        // next 16-col subtile
  v8f acc[2][4];
#pragma unroll
  for (int mi = 0; mi < 2; ++mi)
#pragma unroll
    for (int ni = 0; ni < 4; ++ni) acc[mi][ni] = (v8f){};
  for (int k0 = 0; k0 < K; k0 += 32) {
    if (k0 + 64 < K) {                        // CDNA5 global_prefetch_b8
      __builtin_prefetch(arow0 + k0 + 64, 0, 1);
      __builtin_prefetch(brow + k0 + 64, 0, 1);
    }
    v16bf a0 = load_afrag(arow0 + k0, half);
    v16bf a1 = load_afrag(arow1 + k0, half);
#pragma unroll
    for (int ni = 0; ni < 4; ++ni) {
      v16bf bf_ = *(const v16bf*)(brow + (size_t)ni * bstep + k0 + 16 * half);
      acc[0][ni] = wmma_bf16(a0, bf_, acc[0][ni]);
      acc[1][ni] = wmma_bf16(a1, bf_, acc[1][ni]);
    }
  }
#pragma unroll
  for (int mi = 0; mi < 2; ++mi)
#pragma unroll
    for (int ni = 0; ni < 4; ++ni)
#pragma unroll
      for (int r = 0; r < 8; ++r) {
        int m = m0 + mi * 16 + r + 8 * half;
        size_t o = (size_t)m * N + n0 + ni * 16 + l16;
        float v = acc[mi][ni][r];
        if (Res) v += Res[o];
        C[o] = v;
      }
}

// ---------------------------------------------------------------------------
// Grouped expert GEMM (register-blocked like above) over compact per-expert
// row ranges: rows of expert e live at [offsets[e], offsets[e]+counts[e]).
//   gatherA=1: A row = token rowtok[srow] in A[T][K]; else A compact [TK][K]
//   Wt = [E][N][K] bf16;  C = [TK][N] f32
// grid = (N/128, T/64, E); block 128
// ---------------------------------------------------------------------------
__global__ __launch_bounds__(128) void gemm_group_kernel(
    const bf16* __restrict__ A, const bf16* __restrict__ Wt,
    float* __restrict__ C,
    const int* __restrict__ rowtok, const int* __restrict__ offsets,
    const int* __restrict__ counts, int N, int K, int gatherA) {
  int e  = blockIdx.z;
  int Ce = counts[e];
  int lane = threadIdx.x & 31;
  int wave = threadIdx.x >> 5;
  int half = lane >> 4;
  int l16  = lane & 15;
  int mw = wave & 1, nw = wave >> 1;
  int m0 = blockIdx.y * 64 + mw * 32;         // expert-local row base
  int n0 = blockIdx.x * 128 + nw * 64;
  if (m0 >= Ce || n0 >= N) return;            // wave-uniform exit (EXEC stays full)
  int off = offsets[e];
  int mc0 = m0 + l16;      if (mc0 > Ce - 1) mc0 = Ce - 1;
  int mc1 = m0 + 16 + l16; if (mc1 > Ce - 1) mc1 = Ce - 1;
  const bf16* arow0 = gatherA ? (A + (size_t)rowtok[off + mc0] * K)
                              : (A + (size_t)(off + mc0) * K);
  const bf16* arow1 = gatherA ? (A + (size_t)rowtok[off + mc1] * K)
                              : (A + (size_t)(off + mc1) * K);
  const bf16* brow = Wt + ((size_t)e * N + n0 + l16) * K;
  const size_t bstep = (size_t)16 * K;
  v8f acc[2][4];
#pragma unroll
  for (int mi = 0; mi < 2; ++mi)
#pragma unroll
    for (int ni = 0; ni < 4; ++ni) acc[mi][ni] = (v8f){};
  for (int k0 = 0; k0 < K; k0 += 32) {
    if (k0 + 64 < K) {
      __builtin_prefetch(arow0 + k0 + 64, 0, 1);
      __builtin_prefetch(brow + k0 + 64, 0, 1);
    }
    v16bf a0 = load_afrag(arow0 + k0, half);
    v16bf a1 = load_afrag(arow1 + k0, half);
#pragma unroll
    for (int ni = 0; ni < 4; ++ni) {
      v16bf bf_ = *(const v16bf*)(brow + (size_t)ni * bstep + k0 + 16 * half);
      acc[0][ni] = wmma_bf16(a0, bf_, acc[0][ni]);
      acc[1][ni] = wmma_bf16(a1, bf_, acc[1][ni]);
    }
  }
#pragma unroll
  for (int mi = 0; mi < 2; ++mi)
#pragma unroll
    for (int ni = 0; ni < 4; ++ni)
#pragma unroll
      for (int r = 0; r < 8; ++r) {
        int ml = m0 + mi * 16 + r + 8 * half;
        if (ml < Ce)
          C[((size_t)off + ml) * N + n0 + ni * 16 + l16] = acc[mi][ni][r];
      }
}

// ---------------------------------------------------------------------------
// silu(g) * u : G[rows][2I] f32 -> A1[rows][I] bf16
// ---------------------------------------------------------------------------
__global__ void act_kernel(const float* __restrict__ G, bf16* __restrict__ A1,
                           int rows, int Id, int twoI) {
  size_t i = (size_t)blockIdx.x * 256 + threadIdx.x;
  size_t tot = (size_t)rows * Id;
  if (i >= tot) return;
  size_t r = i / Id, c = i - r * Id;
  float g = G[r * twoI + c];
  float u = G[r * twoI + Id + c];
  float s = g / (1.f + __expf(-g));
  A1[r * Id + c] = f2bf(s * u);
}

// ---------------------------------------------------------------------------
// QKV post-processing: per (token, head): q/k RMSNorm + RoPE -> bf16,
// head-major layouts; V -> transposed [b][kv][dh][s]
// grid = (T, NH+2*NKV), block = 128 (= DH)
// ---------------------------------------------------------------------------
__global__ __launch_bounds__(128) void qkvpost_kernel(
    const float* __restrict__ QKV, const int* __restrict__ pos,
    const float* __restrict__ qw, const float* __restrict__ kw,
    bf16* __restrict__ Qh, bf16* __restrict__ Kh, bf16* __restrict__ Vt) {
  int t  = blockIdx.x;
  int hh = blockIdx.y;
  int b  = t / Sc, s = t % Sc;
  int d  = threadIdx.x;
  const float* row = QKV + (size_t)t * QNc;
  if (hh >= NHc + NKVc) {                    // V path: transpose store only
    int kv = hh - NHc - NKVc;
    float v = row[(NHc + NKVc) * DHc + kv * DHc + d];
    Vt[(((size_t)b * NKVc + kv) * DHc + d) * Sc + s] = f2bf(v);
    return;
  }
  bool isq = hh < NHc;
  int off  = isq ? hh * DHc : (NHc + (hh - NHc)) * DHc;
  float v  = row[off + d];
  __shared__ float red[128];
  __shared__ float sh[DHc];
  red[d] = v * v; __syncthreads();
  for (int st = 64; st > 0; st >>= 1) {
    if (d < st) red[d] += red[d + st];
    __syncthreads();
  }
  float rs = rsqrtf(red[0] / (float)DHc + 1e-6f);
  float xn = v * rs * (isq ? qw[d] : kw[d]);
  sh[d] = xn; __syncthreads();
  int p = pos[s];
  const int half = DHc / 2;                  // 64
  const float LN_THETA = 13.815510557964274f; // ln(1e6)
  float out;
  if (d < half) {
    float inv = __expf(-((float)d / (float)half) * LN_THETA);
    float ang = (float)p * inv;
    out = xn * __cosf(ang) - sh[d + half] * __sinf(ang);
  } else {
    int i = d - half;
    float inv = __expf(-((float)i / (float)half) * LN_THETA);
    float ang = (float)p * inv;
    out = xn * __cosf(ang) + sh[i] * __sinf(ang);
  }
  if (isq) Qh[(((size_t)b * NHc  + hh)        * Sc + s) * DHc + d] = f2bf(out);
  else     Kh[(((size_t)b * NKVc + (hh - NHc)) * Sc + s) * DHc + d] = f2bf(out);
}

// ---------------------------------------------------------------------------
// Flash-style causal GQA attention. One wave per 16-row query tile.
// grid = (S/64, NH, B), block = 128 (4 waves, wave w -> qtile blockIdx.x*4+w)
// Q fragments hoisted out of the kv loop (loop-invariant, 32 VGPRs).
// Scores: WMMA bf16; online softmax via shfl within 16-lane halves;
// P staged through LDS (D-layout -> A-layout); PV via WMMA with V^T.
// ---------------------------------------------------------------------------
__global__ __launch_bounds__(128) void attn_kernel(
    const bf16* __restrict__ Qh, const bf16* __restrict__ Kh,
    const bf16* __restrict__ Vt, bf16* __restrict__ Out) {
  int lane = threadIdx.x & 31;
  int wave = threadIdx.x >> 5;
  int half = lane >> 4;
  int l16  = lane & 15;
  int qt = blockIdx.x * 4 + wave;
  int h  = blockIdx.y;
  int b  = blockIdx.z;
  int q0 = qt * 16;
  int hk = h / (NHc / NKVc);
  const bf16* Qb = Qh + ((size_t)b * NHc  + h ) * Sc * DHc;
  const bf16* Kb = Kh + ((size_t)b * NKVc + hk) * Sc * DHc;
  const bf16* Vb = Vt + ((size_t)b * NKVc + hk) * DHc * Sc;

  __shared__ bf16 ptile[4][16][32];
  bf16 (*P)[32] = ptile[wave];

  // hoist Q fragments (loop-invariant across kv chunks)
  v16bf qf[4];
#pragma unroll
  for (int kk = 0; kk < 4; ++kk)
    qf[kk] = load_afrag(Qb + (size_t)(q0 + l16) * DHc + kk * 32, half);

  float mrow[8], lrow[8];
  v8f o[8];
  v8f vz = {};
#pragma unroll
  for (int r = 0; r < 8; ++r) { mrow[r] = -1e30f; lrow[r] = 0.f; o[r] = vz; }
  const float scale = 0.08838834764831845f;   // DH^-0.5

  for (int kv0 = 0; kv0 < q0 + 16; kv0 += 32) {
    v8f s0 = vz, s1 = vz;
#pragma unroll
    for (int kk = 0; kk < 4; ++kk) {
      v16bf b0 = *(const v16bf*)(Kb + (size_t)(kv0 + l16)      * DHc + kk * 32 + 16 * half);
      v16bf b1 = *(const v16bf*)(Kb + (size_t)(kv0 + 16 + l16) * DHc + kk * 32 + 16 * half);
      s0 = wmma_bf16(qf[kk], b0, s0);
      s1 = wmma_bf16(qf[kk], b1, s1);
    }
    // online softmax update (rows m = r + 8*half, cols across 16-lane half)
#pragma unroll
    for (int r = 0; r < 8; ++r) {
      int m  = r + 8 * half;
      int qi = q0 + m;
      int kj0 = kv0 + l16, kj1 = kv0 + 16 + l16;
      bool ok0 = kj0 <= qi, ok1 = kj1 <= qi;
      float a0 = s0[r] * scale, a1 = s1[r] * scale;
      float mx = fmaxf(ok0 ? a0 : -1e30f, ok1 ? a1 : -1e30f);
      for (int msk = 1; msk < 16; msk <<= 1)
        mx = fmaxf(mx, __shfl_xor(mx, msk, 32));
      float mnew = fmaxf(mrow[r], mx);
      float p0 = ok0 ? __expf(a0 - mnew) : 0.f;
      float p1 = ok1 ? __expf(a1 - mnew) : 0.f;
      float rsum = p0 + p1;
      for (int msk = 1; msk < 16; msk <<= 1)
        rsum += __shfl_xor(rsum, msk, 32);
      float fac = __expf(mrow[r] - mnew);
      lrow[r] = lrow[r] * fac + rsum;
      mrow[r] = mnew;
#pragma unroll
      for (int d = 0; d < 8; ++d) o[d][r] *= fac;
      P[m][l16]      = f2bf(p0);
      P[m][16 + l16] = f2bf(p1);
    }
    // reload P as A fragment (same-wave LDS, in-order)
    v16bf pf = load_afrag(&P[l16][0], half);
#pragma unroll
    for (int d = 0; d < 8; ++d) {
      v16bf bv = *(const v16bf*)(Vb + (size_t)(d * 16 + l16) * Sc + kv0 + 16 * half);
      o[d] = wmma_bf16(pf, bv, o[d]);
    }
  }
#pragma unroll
  for (int r = 0; r < 8; ++r) {
    int m = r + 8 * half;
    float inv = 1.f / lrow[r];
    size_t t = (size_t)b * Sc + q0 + m;
#pragma unroll
    for (int d = 0; d < 8; ++d)
      Out[t * NHDc + h * DHc + d * 16 + l16] = f2bf(o[d][r] * inv);
  }
}

// ---------------------------------------------------------------------------
// Router: logits = x2 @ gate_w, sigmoid, +bias, top-4, normalized weights
// grid = T, block = 128
// ---------------------------------------------------------------------------
__global__ __launch_bounds__(128) void router_kernel(
    const bf16* __restrict__ X2, const float* __restrict__ GW,
    const float* __restrict__ bias, int* __restrict__ topi,
    float* __restrict__ topw, int* __restrict__ counts) {
  int t = blockIdx.x;
  float acc[Ec];
#pragma unroll
  for (int e = 0; e < Ec; ++e) acc[e] = 0.f;
  for (int j = threadIdx.x; j < Hc; j += 128) {
    float xv = bf2f(X2[(size_t)t * Hc + j]);
    const float* g = GW + (size_t)j * Ec;
#pragma unroll
    for (int e = 0; e < Ec; ++e) acc[e] += xv * g[e];
  }
  __shared__ float red[128];
  __shared__ float logit[Ec];
  for (int e = 0; e < Ec; ++e) {
    red[threadIdx.x] = acc[e]; __syncthreads();
    for (int s = 64; s > 0; s >>= 1) {
      if (threadIdx.x < s) red[threadIdx.x] += red[threadIdx.x + s];
      __syncthreads();
    }
    if (threadIdx.x == 0) logit[e] = red[0];
    __syncthreads();
  }
  if (threadIdx.x == 0) {
    float sig[Ec], ch[Ec];
    bool used[Ec];
#pragma unroll
    for (int e = 0; e < Ec; ++e) {
      sig[e] = 1.f / (1.f + __expf(-logit[e]));
      ch[e]  = sig[e] + bias[e];
      used[e] = false;
    }
    int sel[Kc]; float wsum = 0.f;
    for (int k = 0; k < Kc; ++k) {
      int best = 0; float bv = -1e30f;
      for (int e = 0; e < Ec; ++e)
        if (!used[e] && ch[e] > bv) { bv = ch[e]; best = e; }
      used[best] = true; sel[k] = best; wsum += sig[best];
    }
    for (int k = 0; k < Kc; ++k) {
      topi[t * Kc + k] = sel[k];
      topw[t * Kc + k] = sig[sel[k]] / wsum;
      atomicAdd(&counts[sel[k]], 1);
    }
  }
}

__global__ void zero_kernel(int* counts, int* cursor) {
  int i = threadIdx.x;
  if (i < Ec) { counts[i] = 0; cursor[i] = 0; }
}
__global__ void prefix_kernel(const int* __restrict__ counts, int* __restrict__ offsets) {
  if (threadIdx.x == 0) {
    int o = 0;
    for (int e = 0; e < Ec; ++e) { offsets[e] = o; o += counts[e]; }
  }
}
__global__ void scatter_kernel(const int* __restrict__ topi,
                               const int* __restrict__ offsets,
                               int* __restrict__ cursor,
                               int* __restrict__ rowtok, int* __restrict__ slot) {
  int i = blockIdx.x * 256 + threadIdx.x;
  if (i >= TKc) return;
  int e = topi[i];
  int pos = atomicAdd(&cursor[e], 1);
  int s = offsets[e] + pos;
  rowtok[s] = i >> 2;          // token id
  slot[i] = s;
}

// ---------------------------------------------------------------------------
// Final combine: out[t][:] += sum_k w[t][k] * Y[slot(t,k)][:]
// (out already holds residual2 + shared-expert from the previous GEMM)
// ---------------------------------------------------------------------------
__global__ __launch_bounds__(256) void combine_kernel(
    float* __restrict__ Out, const float* __restrict__ Y,
    const int* __restrict__ slot, const float* __restrict__ topw) {
  int t = blockIdx.x;
  int s0 = slot[t * 4 + 0], s1 = slot[t * 4 + 1];
  int s2 = slot[t * 4 + 2], s3 = slot[t * 4 + 3];
  float w0 = topw[t * 4 + 0], w1 = topw[t * 4 + 1];
  float w2 = topw[t * 4 + 2], w3 = topw[t * 4 + 3];
  for (int j = threadIdx.x; j < Hc; j += 256) {
    float v = Out[(size_t)t * Hc + j];
    v += w0 * Y[(size_t)s0 * Hc + j] + w1 * Y[(size_t)s1 * Hc + j]
       + w2 * Y[(size_t)s2 * Hc + j] + w3 * Y[(size_t)s3 * Hc + j];
    Out[(size_t)t * Hc + j] = v;
  }
}

// ---------------------------------------------------------------------------
// Host-side orchestration
// ---------------------------------------------------------------------------
extern "C" void kernel_launch(void* const* d_in, const int* in_sizes, int n_in,
                              void* d_out, int out_size, void* d_ws, size_t ws_size,
                              hipStream_t stream) {
  (void)in_sizes; (void)n_in; (void)out_size; (void)ws_size;

  const float* hidden  = (const float*)d_in[0];
  const int*   pos     = (const int*)  d_in[1];
  const float* ln1_w   = (const float*)d_in[2];
  const float* ln2_w   = (const float*)d_in[3];
  const float* w_qkv   = (const float*)d_in[4];
  const float* w_o     = (const float*)d_in[5];
  const float* q_nw    = (const float*)d_in[6];
  const float* k_nw    = (const float*)d_in[7];
  const float* gate_w  = (const float*)d_in[8];
  const float* c_bias  = (const float*)d_in[9];
  const float* w_gu    = (const float*)d_in[10];
  const float* w_dn    = (const float*)d_in[11];
  const float* ws_gu   = (const float*)d_in[12];
  const float* ws_dn   = (const float*)d_in[13];
  float* out = (float*)d_out;

  char* p = (char*)d_ws;
  auto take = [&](size_t bytes) -> char* {
    char* r = p;
    p += (bytes + 255) & ~(size_t)255;
    return r;
  };
  bf16*  wqkv_t = (bf16*) take((size_t)QNc * Hc * 2);
  bf16*  wo_t   = (bf16*) take((size_t)Hc * NHDc * 2);
  bf16*  wgu_t  = (bf16*) take((size_t)Ec * I2c * Hc * 2);
  bf16*  wdn_t  = (bf16*) take((size_t)Ec * Hc * Ic * 2);
  bf16*  wsgu_t = (bf16*) take((size_t)ISH2c * Hc * 2);
  bf16*  wsdn_t = (bf16*) take((size_t)Hc * ISHc * 2);
  bf16*  xb     = (bf16*) take((size_t)Tc * Hc * 2);
  float* qkv    = (float*)take((size_t)Tc * QNc * 4);
  bf16*  qh     = (bf16*) take((size_t)Bc * NHc * Sc * DHc * 2);
  bf16*  kh     = (bf16*) take((size_t)Bc * NKVc * Sc * DHc * 2);
  bf16*  vT     = (bf16*) take((size_t)Bc * NKVc * DHc * Sc * 2);
  bf16*  attn   = (bf16*) take((size_t)Tc * NHDc * 2);
  float* hbuf   = (float*)take((size_t)Tc * Hc * 4);
  bf16*  x2b    = (bf16*) take((size_t)Tc * Hc * 2);
  float* G      = (float*)take((size_t)TKc * I2c * 4);
  bf16*  a1     = (bf16*) take((size_t)TKc * Ic * 2);
  float* Y      = (float*)take((size_t)TKc * Hc * 4);
  float* Gs     = (float*)take((size_t)Tc * ISH2c * 4);
  bf16*  as1    = (bf16*) take((size_t)Tc * ISHc * 2);
  int*   topi   = (int*)  take((size_t)TKc * 4);
  float* topw   = (float*)take((size_t)TKc * 4);
  int*   counts = (int*)  take(64);
  int*   offs   = (int*)  take(64);
  int*   cursor = (int*)  take(64);
  int*   rowtok = (int*)  take((size_t)TKc * 4);
  int*   slot   = (int*)  take((size_t)TKc * 4);

  auto cblocks = [](size_t tot) { return (unsigned)((tot + 255) / 256); };

  // --- weight conversion (f32 -> bf16, transposed to [N][K]) ---
  conv_t_kernel<<<cblocks((size_t)Hc * QNc),        256, 0, stream>>>(w_qkv, wqkv_t, Hc,  QNc, 1);
  conv_t_kernel<<<cblocks((size_t)NHDc * Hc),       256, 0, stream>>>(w_o,   wo_t,   NHDc, Hc, 1);
  conv_t_kernel<<<cblocks((size_t)Ec * Hc * I2c),   256, 0, stream>>>(w_gu,  wgu_t,  Hc,  I2c, Ec);
  conv_t_kernel<<<cblocks((size_t)Ec * Ic * Hc),    256, 0, stream>>>(w_dn,  wdn_t,  Ic,  Hc,  Ec);
  conv_t_kernel<<<cblocks((size_t)Hc * ISH2c),      256, 0, stream>>>(ws_gu, wsgu_t, Hc,  ISH2c, 1);
  conv_t_kernel<<<cblocks((size_t)ISHc * Hc),       256, 0, stream>>>(ws_dn, wsdn_t, ISHc, Hc, 1);

  // --- attention block ---
  rmsnorm_kernel<<<Tc, 256, 0, stream>>>(hidden, ln1_w, xb, Hc);
  gemm_bf16_kernel<<<dim3(QNc / 128, Tc / 64), 128, 0, stream>>>(
      xb, wqkv_t, qkv, nullptr, Tc, QNc, Hc);
  qkvpost_kernel<<<dim3(Tc, NHc + 2 * NKVc), 128, 0, stream>>>(
      qkv, pos, q_nw, k_nw, qh, kh, vT);
  attn_kernel<<<dim3(Sc / 64, NHc, Bc), 128, 0, stream>>>(qh, kh, vT, attn);
  gemm_bf16_kernel<<<dim3(Hc / 128, Tc / 64), 128, 0, stream>>>(
      attn, wo_t, hbuf, hidden, Tc, Hc, NHDc);

  // --- MoE routing ---
  rmsnorm_kernel<<<Tc, 256, 0, stream>>>(hbuf, ln2_w, x2b, Hc);
  zero_kernel<<<1, 32, 0, stream>>>(counts, cursor);
  router_kernel<<<Tc, 128, 0, stream>>>(x2b, gate_w, c_bias, topi, topw, counts);
  prefix_kernel<<<1, 32, 0, stream>>>(counts, offs);
  scatter_kernel<<<(TKc + 255) / 256, 256, 0, stream>>>(topi, offs, cursor, rowtok, slot);

  // --- routed experts (grouped WMMA GEMMs on compact rows) ---
  gemm_group_kernel<<<dim3(I2c / 128, Tc / 64, Ec), 128, 0, stream>>>(
      x2b, wgu_t, G, rowtok, offs, counts, I2c, Hc, 1);
  act_kernel<<<cblocks((size_t)TKc * Ic), 256, 0, stream>>>(G, a1, TKc, Ic, I2c);
  gemm_group_kernel<<<dim3(Hc / 128, Tc / 64, Ec), 128, 0, stream>>>(
      a1, wdn_t, Y, rowtok, offs, counts, Hc, Ic, 0);

  // --- shared expert (out = residual2 + shared) ---
  gemm_bf16_kernel<<<dim3(ISH2c / 128, Tc / 64), 128, 0, stream>>>(
      x2b, wsgu_t, Gs, nullptr, Tc, ISH2c, Hc);
  act_kernel<<<cblocks((size_t)Tc * ISHc), 256, 0, stream>>>(Gs, as1, Tc, ISHc, ISH2c);
  gemm_bf16_kernel<<<dim3(Hc / 128, Tc / 64), 128, 0, stream>>>(
      as1, wsdn_t, out, hbuf, Tc, Hc, ISHc);

  // --- out += routed ---
  combine_kernel<<<Tc, 256, 0, stream>>>(out, Y, slot, topw);
}
